// MixtureAttention_20478404067608
// MI455X (gfx1250) — compile-verified, matrix operand order checked by reading, not run on hardware
//
#include <hip/hip_runtime.h>
#include <hip/hip_bf16.h>

// ---------------------------------------------------------------------------
// MixtureAttention for MI455X (gfx1250, wave32, WMMA + Tensor Data Mover).
// All matmuls: v_wmma_f32_16x16x32_bf16 (f32 accumulate).
// Operand tiles staged in LDS via TDM (tensor_load_to_lds), TRIPLE-buffered
// with prefetch distance 2; descriptor issue split across wave0/wave1 so each
// issuing wave tracks one in-order TDM stream on its own TENSORcnt.
// Attention: flash-style online softmax, scores never hit HBM.
// ---------------------------------------------------------------------------

#define D_MODEL 1024
#define NHEAD   16
#define HDIM    64
#define SEQ     2048
#define BATCH   2
#define MTOT    (BATCH * SEQ)     // 4096 rows
#define MAXL    5000

typedef __bf16 bf16_t;
typedef bf16_t v16bf __attribute__((ext_vector_type(16)));
typedef bf16_t v8bf  __attribute__((ext_vector_type(8)));
typedef float  v8f   __attribute__((ext_vector_type(8)));
typedef unsigned int v4u __attribute__((ext_vector_type(4)));
typedef int v4i __attribute__((ext_vector_type(4)));
typedef int v8i __attribute__((ext_vector_type(8)));

__device__ __forceinline__ v8f wmma_bf16(v16bf a, v16bf b, v8f c) {
  return __builtin_amdgcn_wmma_f32_16x16x32_bf16(false, a, false, b, (short)0, c,
                                                 false, false);
}

// Load a 16x32 bf16 fragment in A layout (also B layout when the source holds
// B^T row-major): lane l -> row r0+(l&15); K block k0+(l>>4)*8 and +16.
// Two 16-byte loads per lane (global_load_b128 or ds_load_b128).
__device__ __forceinline__ v16bf load_frag(const bf16_t* __restrict__ src, int ld,
                                           int r0, int k0, int lane) {
  const int r  = r0 + (lane & 15);
  const int kb = k0 + ((lane >> 4) << 3);
  const bf16_t* p = src + (size_t)r * ld + kb;
  v8bf lo = *(const v8bf*)p;
  v8bf hi = *(const v8bf*)(p + 16);
  v16bf a;
#pragma unroll
  for (int e = 0; e < 8; ++e) { a[e] = lo[e]; a[8 + e] = hi[e]; }
  return a;
}

// ---------------------------------------------------------------------------
// TDM: DMA a 2-D bf16 tile (tile_d0 contiguous elems x tile_d1 rows, row
// stride stride_elems) from global into LDS (packed, ld = tile_d0).
// D# packing per CDNA5 ISA 8.3/8.4: count=1, type=2, data_size=1 (2 bytes),
// workgroup_mask=0 (not in a cluster), no padding / iteration / gather.
// ---------------------------------------------------------------------------
__device__ __forceinline__ void tdm_load_2d(void* lds_dst, const void* gsrc,
                                            unsigned tile_d0, unsigned tile_d1,
                                            unsigned stride_elems) {
  const unsigned long long ga = (unsigned long long)gsrc;
  const unsigned lds = (unsigned)(unsigned long long)lds_dst; // addr[31:0] = LDS offset

  v4u g0;
  g0[0] = 1u;                                            // count=1
  g0[1] = lds;                                           // lds_addr
  g0[2] = (unsigned)(ga & 0xFFFFFFFFull);                // global_addr[31:0]
  g0[3] = (unsigned)((ga >> 32) & 0x1FFFFFFull) | (2u << 30); // addr[56:32] | type=2

  v8i g1;
  g1[0] = (int)(1u << 16);                               // data_size=1 (2B)
  g1[1] = (int)((tile_d0 & 0xFFFFu) << 16);              // tensor_dim0[15:0]
  g1[2] = (int)(((tile_d0 >> 16) & 0xFFFFu) |
                ((tile_d1 & 0xFFFFu) << 16));            // dim0 hi | tensor_dim1 lo
  g1[3] = (int)(((tile_d1 >> 16) & 0xFFFFu) |
                ((tile_d0 & 0xFFFFu) << 16));            // dim1 hi | tile_dim0
  g1[4] = (int)(tile_d1 & 0xFFFFu);                      // tile_dim1 (tile_dim2=0)
  g1[5] = (int)stride_elems;                             // tensor_dim0_stride[31:0]
  g1[6] = 0;                                             // stride hi | dim1_stride lo
  g1[7] = 0;

  v4i z4; z4[0] = z4[1] = z4[2] = z4[3] = 0;
#if __clang_major__ >= 23
  v8i z8;
#pragma unroll
  for (int i = 0; i < 8; ++i) z8[i] = 0;
  __builtin_amdgcn_tensor_load_to_lds(g0, g1, z4, z4, z8, 0);
#else
  __builtin_amdgcn_tensor_load_to_lds(g0, g1, z4, z4, 0);
#endif
}

// xor-reductions over the 16-lane half a row lives in (masks 1,2,4,8 stay
// inside each half of the wave32).
__device__ __forceinline__ float half16_max(float v) {
  v = fmaxf(v, __int_as_float(__builtin_amdgcn_ds_swizzle(__float_as_int(v), 0x041F)));
  v = fmaxf(v, __int_as_float(__builtin_amdgcn_ds_swizzle(__float_as_int(v), 0x081F)));
  v = fmaxf(v, __int_as_float(__builtin_amdgcn_ds_swizzle(__float_as_int(v), 0x101F)));
  v = fmaxf(v, __int_as_float(__builtin_amdgcn_ds_swizzle(__float_as_int(v), 0x201F)));
  return v;
}
__device__ __forceinline__ float half16_sum(float v) {
  v += __int_as_float(__builtin_amdgcn_ds_swizzle(__float_as_int(v), 0x041F));
  v += __int_as_float(__builtin_amdgcn_ds_swizzle(__float_as_int(v), 0x081F));
  v += __int_as_float(__builtin_amdgcn_ds_swizzle(__float_as_int(v), 0x101F));
  v += __int_as_float(__builtin_amdgcn_ds_swizzle(__float_as_int(v), 0x201F));
  return v;
}

// ---------------------------------------------------------------------------
// 1) x + mixed positional encoding -> bf16 [MTOT, D]
// ---------------------------------------------------------------------------
__global__ void pe_add_kernel(const float* __restrict__ x,
                              const float* __restrict__ rel,
                              const float* __restrict__ alpha_p,
                              bf16_t* __restrict__ xe) {
  const int idx = blockIdx.x * blockDim.x + threadIdx.x;   // [0, MTOT*D)
  const int d   = idx & (D_MODEL - 1);
  const int row = idx >> 10;           // b*SEQ + s
  const int s   = row & (SEQ - 1);
  const float alpha = alpha_p[0];
  const int   i    = d >> 1;
  const float freq = __expf((float)(2 * i) * (-9.210340371976184f / (float)D_MODEL));
  const float ang  = (float)s * freq;
  const float pe_a = (d & 1) ? __cosf(ang) : __sinf(ang);
  const float pe_r = rel[(size_t)(MAXL - SEQ + s) * D_MODEL + d];
  const float v = x[idx] + alpha * pe_a + (1.0f - alpha) * pe_r;
  xe[idx] = (bf16_t)v;
}

// ---------------------------------------------------------------------------
// 2) f32 -> bf16 weight conversion
// ---------------------------------------------------------------------------
__global__ void cvt_bf16_kernel(const float* __restrict__ w, bf16_t* __restrict__ o) {
  const int i = blockIdx.x * blockDim.x + threadIdx.x;
  o[i] = (bf16_t)w[i];
}

// ---------------------------------------------------------------------------
// 3) GEMM: C[m,n] = sum_k A[m,k] * W[n,k] + bias[n]  (y = x W^T + b)
//    K-step 64; A (64x64) + W (64x64) tiles TDM-staged, 3 buffers, prefetch
//    distance 2.  wave0 issues A tiles, wave1 issues W tiles.
//    block = 128 thr (4 waves); wave -> 16(M) x 64(N); grid (M/64, N/64)
//    mode 0: bf16 [B,H,S,64] (Q,K) | mode 1: bf16 [B,H,64,S] (V^T)
//    mode 2: f32 [M, D] + bias (output projection)
// ---------------------------------------------------------------------------
#define GK 64
#define GNIT (D_MODEL / GK)   // 16

__global__ void gemm_kernel(const bf16_t* __restrict__ A,
                            const bf16_t* __restrict__ W,
                            const float* __restrict__ bias,
                            bf16_t* __restrict__ outb,
                            float* __restrict__ outf, int mode) {
  __shared__ bf16_t As[3][64 * GK];
  __shared__ bf16_t Bs[3][64 * GK];
  const int lane = threadIdx.x & 31;
  const int wid  = threadIdx.x >> 5;
  const int mblk = blockIdx.x * 64;
  const int n0   = blockIdx.y * 64;
  const int m0   = mblk + wid * 16;
  const bf16_t* Ab = A + (size_t)mblk * D_MODEL;
  const bf16_t* Wb = W + (size_t)n0 * D_MODEL;

  v8f c[4];
#pragma unroll
  for (int i = 0; i < 4; ++i)
#pragma unroll
    for (int e = 0; e < 8; ++e) c[i][e] = 0.0f;

  // prologue: tiles for it=0,1
  if (wid == 0) { tdm_load_2d(As[0], Ab, GK, 64, D_MODEL);
                  tdm_load_2d(As[1], Ab + GK, GK, 64, D_MODEL); }
  else if (wid == 1) { tdm_load_2d(Bs[0], Wb, GK, 64, D_MODEL);
                       tdm_load_2d(Bs[1], Wb + GK, GK, 64, D_MODEL); }

  int cur = 0;
  for (int it = 0; it < GNIT; ++it) {
    if (wid < 2) {
      if (it + 1 < GNIT) __builtin_amdgcn_s_wait_tensorcnt(1);  // tile it done
      else               __builtin_amdgcn_s_wait_tensorcnt(0);
    }
    __syncthreads();     // publish buf[cur]; all reads of buf[cur-1] finished
    if (it + 2 < GNIT) {
      const int nb = (cur >= 1) ? cur - 1 : 2;   // (cur+2) % 3
      const int kf = (it + 2) * GK;
      if (wid == 0)      tdm_load_2d(As[nb], Ab + kf, GK, 64, D_MODEL);
      else if (wid == 1) tdm_load_2d(Bs[nb], Wb + kf, GK, 64, D_MODEL);
    }

    // batched fragment loads (distinct regs) -> one DS clause, staggered waits
    const v16bf a0 = load_frag(As[cur], GK, wid * 16, 0,  lane);
    const v16bf a1 = load_frag(As[cur], GK, wid * 16, 32, lane);
    v16bf bl[4], bh[4];
#pragma unroll
    for (int i = 0; i < 4; ++i) {
      bl[i] = load_frag(Bs[cur], GK, i * 16, 0,  lane);
      bh[i] = load_frag(Bs[cur], GK, i * 16, 32, lane);
    }
#pragma unroll
    for (int i = 0; i < 4; ++i) {
      c[i] = wmma_bf16(a0, bl[i], c[i]);
      c[i] = wmma_bf16(a1, bh[i], c[i]);
    }
    cur = (cur + 1 < 3) ? cur + 1 : 0;
  }

  const int half = lane >> 4, nl = lane & 15;
#pragma unroll
  for (int i = 0; i < 4; ++i) {
    const int n  = n0 + i * 16 + nl;
    const float bv = bias[n];
    const int h = n >> 6, dc = n & 63;
#pragma unroll
    for (int r = 0; r < 8; ++r) {
      const int m = m0 + half * 8 + r;
      const float v = c[i][r] + bv;
      if (mode == 2) {
        outf[(size_t)m * D_MODEL + n] = v;
      } else {
        const int b = m >> 11, s = m & (SEQ - 1);
        const size_t bh = (size_t)(b * NHEAD + h);
        const size_t o = (mode == 0) ? (bh * SEQ + s) * HDIM + dc
                                     : (bh * HDIM + dc) * SEQ + s;
        outb[o] = (bf16_t)v;
      }
    }
  }
}

// ---------------------------------------------------------------------------
// 4) Flash attention.  grid (SEQ/64, B*H); block 128 (4 waves).
//    wave: 16 query rows; 32 keys per step.  K (32x64) / V^T (64x32) tiles
//    TDM-staged, 3 buffers, prefetch distance 2; wave0 issues K, wave1 V.
//    scale = 1/sqrt(D_MODEL) = 1/32 (reference scales by d_model!).
// ---------------------------------------------------------------------------
#define FNIT (SEQ / 32)   // 64

__global__ void flash_kernel(const bf16_t* __restrict__ Q,
                             const bf16_t* __restrict__ K,
                             const bf16_t* __restrict__ VT,
                             bf16_t* __restrict__ att) {
  __shared__ bf16_t Ks[3][32 * 64];
  __shared__ bf16_t Vs[3][64 * 32];
  __shared__ bf16_t pshm[4][16 * 32];     // per-wave P tile (16x32 bf16)
  const int lane = threadIdx.x & 31;
  const int wid  = threadIdx.x >> 5;
  const int bh = blockIdx.y;
  const int b = bh >> 4, h = bh & 15;
  const bf16_t* qh = Q  + (size_t)bh * SEQ * HDIM;
  const bf16_t* kh = K  + (size_t)bh * SEQ * HDIM;
  const bf16_t* vh = VT + (size_t)bh * HDIM * SEQ;
  const int m0 = blockIdx.x * 64 + wid * 16;   // query row base (within seq)
  const int half = lane >> 4, nl = lane & 15;
  bf16_t* pl = pshm[wid];

  const v16bf qa0 = load_frag(qh, HDIM, m0, 0,  lane);
  const v16bf qa1 = load_frag(qh, HDIM, m0, 32, lane);

  v8f acc[4];
  float rmax[8], rsum[8];
#pragma unroll
  for (int i = 0; i < 4; ++i)
#pragma unroll
    for (int e = 0; e < 8; ++e) acc[i][e] = 0.0f;
#pragma unroll
  for (int r = 0; r < 8; ++r) { rmax[r] = -3.0e38f; rsum[r] = 0.0f; }

  const float scale = 0.03125f;  // 1/sqrt(1024)

  if (wid == 0) { tdm_load_2d(Ks[0], kh, 64, 32, HDIM);
                  tdm_load_2d(Ks[1], kh + 32 * HDIM, 64, 32, HDIM); }
  else if (wid == 1) { tdm_load_2d(Vs[0], vh, 32, 64, SEQ);
                       tdm_load_2d(Vs[1], vh + 32, 32, 64, SEQ); }

  int cur = 0;
  for (int it = 0; it < FNIT; ++it) {
    if (wid < 2) {
      if (it + 1 < FNIT) __builtin_amdgcn_s_wait_tensorcnt(1);
      else               __builtin_amdgcn_s_wait_tensorcnt(0);
    }
    __syncthreads();
    if (it + 2 < FNIT) {
      const int nb = (cur >= 1) ? cur - 1 : 2;   // (cur+2) % 3
      const int kf = (it + 2) * 32;
      if (wid == 0)      tdm_load_2d(Ks[nb], kh + (size_t)kf * HDIM, 64, 32, HDIM);
      else if (wid == 1) tdm_load_2d(Vs[nb], vh + kf, 32, 64, SEQ);
    }

    // ---- scores for 32 keys: two 16x16 D tiles; batched K-frag loads ----
    const v16bf kb00 = load_frag(Ks[cur], HDIM, 0,  0,  lane);
    const v16bf kb01 = load_frag(Ks[cur], HDIM, 0,  32, lane);
    const v16bf kb10 = load_frag(Ks[cur], HDIM, 16, 0,  lane);
    const v16bf kb11 = load_frag(Ks[cur], HDIM, 16, 32, lane);
    v8f s0, s1;
#pragma unroll
    for (int e = 0; e < 8; ++e) { s0[e] = 0.0f; s1[e] = 0.0f; }
    s0 = wmma_bf16(qa0, kb00, s0);
    s0 = wmma_bf16(qa1, kb01, s0);
    s1 = wmma_bf16(qa0, kb10, s1);
    s1 = wmma_bf16(qa1, kb11, s1);

    // ---- online softmax over this 32-key tile ----
#pragma unroll
    for (int r = 0; r < 8; ++r) {
      const float a0 = s0[r] * scale;
      const float a1 = s1[r] * scale;
      const float tmax = half16_max(fmaxf(a0, a1));
      const float nm   = fmaxf(rmax[r], tmax);
      const float corr = __expf(rmax[r] - nm);
      rmax[r] = nm;
      const float p0 = __expf(a0 - nm);
      const float p1 = __expf(a1 - nm);
      rsum[r] = rsum[r] * corr + half16_sum(p0 + p1);
#pragma unroll
      for (int i = 0; i < 4; ++i) acc[i][r] = acc[i][r] * corr;
      // D layout -> LDS (row-major 16x32 bf16) for A-layout reload
      pl[(half * 8 + r) * 32 + nl]      = (bf16_t)p0;
      pl[(half * 8 + r) * 32 + 16 + nl] = (bf16_t)p1;
    }
    asm volatile("s_wait_dscnt 0x0" ::: "memory");   // intra-wave LDS RAW

    // ---- P (16x32) x V (32x64): batched V-frag loads, 4 WMMA ----
    const v16bf pa = load_frag(pl, 32, 0, 0, lane);
    v16bf vb[4];
#pragma unroll
    for (int i = 0; i < 4; ++i) vb[i] = load_frag(Vs[cur], 32, i * 16, 0, lane);
#pragma unroll
    for (int i = 0; i < 4; ++i) acc[i] = wmma_bf16(pa, vb[i], acc[i]);

    cur = (cur + 1 < 3) ? cur + 1 : 0;
  }

  // ---- normalize and write head-concat layout [B*S, D] bf16 ----
#pragma unroll
  for (int r = 0; r < 8; ++r) {
    const float inv = 1.0f / rsum[r];
    const int s = m0 + half * 8 + r;
    const size_t row = (size_t)b * SEQ + s;
#pragma unroll
    for (int i = 0; i < 4; ++i) {
      const int col = h * HDIM + i * 16 + nl;
      att[row * D_MODEL + col] = (bf16_t)(acc[i][r] * inv);
    }
  }
}

// ---------------------------------------------------------------------------
extern "C" void kernel_launch(void* const* d_in, const int* in_sizes, int n_in,
                              void* d_out, int out_size, void* d_ws, size_t ws_size,
                              hipStream_t stream) {
  const float* x     = (const float*)d_in[0];
  const float* rel   = (const float*)d_in[1];
  const float* alpha = (const float*)d_in[2];
  const float* Wq = (const float*)d_in[3];
  const float* bq = (const float*)d_in[4];
  const float* Wk = (const float*)d_in[5];
  const float* bk = (const float*)d_in[6];
  const float* Wv = (const float*)d_in[7];
  const float* bv = (const float*)d_in[8];
  const float* Wo = (const float*)d_in[9];
  const float* bo = (const float*)d_in[10];
  float* out = (float*)d_out;

  // workspace carve-up (bf16 elements); total = 24M elems = 48 MB
  bf16_t* p   = (bf16_t*)d_ws;
  bf16_t* xe  = p;                 p += (size_t)MTOT * D_MODEL;     // 4M
  bf16_t* wqb = p;                 p += (size_t)D_MODEL * D_MODEL;  // 1M each
  bf16_t* wkb = p;                 p += (size_t)D_MODEL * D_MODEL;
  bf16_t* wvb = p;                 p += (size_t)D_MODEL * D_MODEL;
  bf16_t* wob = p;                 p += (size_t)D_MODEL * D_MODEL;
  bf16_t* qb  = p;                 p += (size_t)MTOT * D_MODEL;     // [B,H,S,64]
  bf16_t* kb  = p;                 p += (size_t)MTOT * D_MODEL;     // [B,H,S,64]
  bf16_t* vtb = p;                 p += (size_t)MTOT * D_MODEL;     // [B,H,64,S]
  bf16_t* att = p;                 p += (size_t)MTOT * D_MODEL;     // [B*S, D]

  // 1) PE fuse + bf16 cast
  pe_add_kernel<<<(MTOT * D_MODEL) / 256, 256, 0, stream>>>(x, rel, alpha, xe);

  // 2) weight conversion
  const int wblk = (D_MODEL * D_MODEL) / 256;
  cvt_bf16_kernel<<<wblk, 256, 0, stream>>>(Wq, wqb);
  cvt_bf16_kernel<<<wblk, 256, 0, stream>>>(Wk, wkb);
  cvt_bf16_kernel<<<wblk, 256, 0, stream>>>(Wv, wvb);
  cvt_bf16_kernel<<<wblk, 256, 0, stream>>>(Wo, wob);

  // 3) Q,K,V projections (TDM-staged GEMM)
  dim3 ggrid(MTOT / 64, D_MODEL / 64);
  gemm_kernel<<<ggrid, 128, 0, stream>>>(xe, wqb, bq, qb, nullptr, 0);
  gemm_kernel<<<ggrid, 128, 0, stream>>>(xe, wkb, bk, kb, nullptr, 0);
  gemm_kernel<<<ggrid, 128, 0, stream>>>(xe, wvb, bv, vtb, nullptr, 1);

  // 4) flash attention (TDM-staged K/V tiles)
  flash_kernel<<<dim3(SEQ / 64, BATCH * NHEAD), 128, 0, stream>>>(qb, kb, vtb, att);

  // 5) output projection
  gemm_kernel<<<ggrid, 128, 0, stream>>>(att, wob, bo, nullptr, out, 2);
}
